// TCAttentionBlock_10565619548682
// MI455X (gfx1250) — matrix operand order repaired and checked by reading, not compiled, not run on hardware
//
#include <hip/hip_runtime.h>
#include <hip/hip_bf16.h>
#include <float.h>

// ---------------- problem constants ----------------
#define D_     768
#define H_     12
#define T_     16
#define NP_    196
#define N_     197
#define TN_    3152          // T_*N_
#define HD_    64
#define TOPS_  58
#define KCTX_  96
#define B_     8
#define L_     3248          // TN_+KCTX_
#define BT_    128           // B_*T_
#define BL_    (B_*L_)       // 25984
#define KEYS_  293           // N_+KCTX_
#define KP_    320           // padded keys (10 chunks of 32)
#define KCH_   10            // KP_/32
#define QTILES_ 13           // ceil(197/16)
#define FF_    3072
#define SEED0_ 928           // TOPS_*T_
#define NA_MAX_ 464

typedef _Float16 f16;
typedef __attribute__((ext_vector_type(16))) _Float16 v16h;
typedef __attribute__((ext_vector_type(8)))  float    v8f;

static __device__ inline v8f wmma32(v16h a, v16h b, v8f c) {
  // D = A(16x32 f16) * B(32x16 f16) + C(16x16 f32)
  return __builtin_amdgcn_wmma_f32_16x16x32_f16(false, a, false, b, (short)0, c,
                                                false, false);
}

// ---------------- layernorm -> f16 ----------------
__global__ void ln_f16_kernel(const float* __restrict__ x, const float* __restrict__ g,
                              const float* __restrict__ bb, f16* __restrict__ out,
                              int rows) {
  int row = blockIdx.x;
  if (row >= rows) return;
  const float* xr = x + (size_t)row * D_;
  __shared__ float sd[256];
  int tid = threadIdx.x;
  float s = 0.f, sq = 0.f;
  for (int c = tid; c < D_; c += 256) { float v = xr[c]; s += v; sq += v * v; }
  sd[tid] = s; __syncthreads();
  for (int o = 128; o > 0; o >>= 1) { if (tid < o) sd[tid] += sd[tid + o]; __syncthreads(); }
  float mean = sd[0] * (1.f / D_);
  __syncthreads();
  sd[tid] = sq; __syncthreads();
  for (int o = 128; o > 0; o >>= 1) { if (tid < o) sd[tid] += sd[tid + o]; __syncthreads(); }
  float var = sd[0] * (1.f / D_) - mean * mean;
  float inv = rsqrtf(var + 1e-5f);
  for (int c = tid; c < D_; c += 256) {
    float v = (xr[c] - mean) * inv * g[c] + bb[c];
    out[(size_t)row * D_ + c] = (f16)v;
  }
}

// ---------------- pack weight f32[K][Nn] -> f16 B-fragment layout ----------------
// Wp[((kt*Nn + n)*32) + kk] = (f16)W[(kt*32+kk)*Nn + n]
// Per-lane fragment (col n, lane-half hh): 16 contiguous f16 at (kt*Nn+n)*32 + hh*16
__global__ void pack_w_kernel(const float* __restrict__ W, f16* __restrict__ Wp,
                              int K, int Nn) {
  size_t i = (size_t)blockIdx.x * 256 + threadIdx.x;
  size_t total = (size_t)K * Nn;
  if (i >= total) return;
  int kk = (int)(i & 31);
  size_t rest = i >> 5;
  int n  = (int)(rest % Nn);
  int kt = (int)(rest / Nn);
  Wp[i] = (f16)W[(size_t)(kt * 32 + kk) * Nn + n];
}

// ---------------- pack V (f16) into per-(bt,h,chunk) B-fragment layout ----------------
// Vp[(((bt*H + h)*KCH + ch)*64 + hd)*32 + kk] = v16[rowmap(b,t,ch*32+kk)][h*64+hd] (0 if key>=KEYS_)
__global__ void pack_v_kernel(const f16* __restrict__ v16, f16* __restrict__ Vp) {
  size_t i = (size_t)blockIdx.x * 256 + threadIdx.x;
  size_t total = (size_t)BT_ * H_ * KCH_ * 64 * 32;
  if (i >= total) return;
  int kk = (int)(i & 31);
  size_t i2 = i >> 5;
  int hd = (int)(i2 & 63);
  size_t i3 = i2 >> 6;
  int ch = (int)(i3 % KCH_);
  size_t i4 = i3 / KCH_;
  int h  = (int)(i4 % H_);
  int bt = (int)(i4 / H_);
  int b = bt / T_, t = bt % T_;
  int key = ch * 32 + kk;
  f16 val = (f16)0.f;
  if (key < KEYS_) {
    size_t row = (size_t)b * L_ + (key < N_ ? (size_t)t * N_ + key : (size_t)TN_ + (key - N_));
    val = v16[row * D_ + h * HD_ + hd];
  }
  Vp[i] = val;
}

// ---------------- generic WMMA GEMM (packed B) ----------------
// C[M,Nn] = A[M,K] (f16 row-major) @ B (f16 packed fragment layout) + bias
// Block: 128 threads = 4 waves; block tile 16 x 256; each wave: 16 x 64 (4 accumulators).
// act: 0=none, 1=quick_gelu.  rowremap: 0=identity, 1: row -> (row/TN_)*L_ + row%TN_
__global__ void gemm_wmma_kernel(const f16* __restrict__ A, const f16* __restrict__ Bp,
                                 const float* __restrict__ bias,
                                 const float* __restrict__ Res,
                                 float* __restrict__ Of32, f16* __restrict__ Of16,
                                 int M, int Nn, int K, int act, int rowremap) {
  int lane = threadIdx.x & 31;
  int wv   = threadIdx.x >> 5;            // 0..3
  int c0   = blockIdx.x * 256 + wv * 64;
  int r0   = blockIdx.y * 16;
  int hh   = lane >> 4;
  int rA   = r0 + (lane & 15); if (rA >= M) rA = M - 1;

  v8f acc0 = {}, acc1 = {}, acc2 = {}, acc3 = {};
  const f16* pa = A + (size_t)rA * K;
  for (int k0 = 0; k0 < K; k0 += 32) {
    v16h a;
#pragma unroll
    for (int e = 0; e < 8; e++) a[e]     = pa[k0 + hh * 16 + e];
#pragma unroll
    for (int e = 0; e < 8; e++) a[8 + e] = pa[k0 + hh * 16 + 8 + e];
    __builtin_prefetch(pa + k0 + 64, 0, 0);
    size_t ktbase = (size_t)(k0 >> 5) * Nn;
    v16h b0, b1, b2, b3;
    {
      const f16* pb = Bp + (ktbase + c0 +  0 + (lane & 15)) * 32 + hh * 16;
#pragma unroll
      for (int e = 0; e < 16; e++) b0[e] = pb[e];
    }
    {
      const f16* pb = Bp + (ktbase + c0 + 16 + (lane & 15)) * 32 + hh * 16;
#pragma unroll
      for (int e = 0; e < 16; e++) b1[e] = pb[e];
    }
    {
      const f16* pb = Bp + (ktbase + c0 + 32 + (lane & 15)) * 32 + hh * 16;
#pragma unroll
      for (int e = 0; e < 16; e++) b2[e] = pb[e];
    }
    {
      const f16* pb = Bp + (ktbase + c0 + 48 + (lane & 15)) * 32 + hh * 16;
#pragma unroll
      for (int e = 0; e < 16; e++) b3[e] = pb[e];
    }
    acc0 = wmma32(a, b0, acc0);
    acc1 = wmma32(a, b1, acc1);
    acc2 = wmma32(a, b2, acc2);
    acc3 = wmma32(a, b3, acc3);
  }
  __builtin_amdgcn_s_wait_tensorcnt(0);   // CDNA5 split-counter path (TENSORcnt==0, no-op)

#pragma unroll
  for (int u = 0; u < 4; u++) {
    v8f acc = (u == 0) ? acc0 : (u == 1) ? acc1 : (u == 2) ? acc2 : acc3;
    int cn = c0 + u * 16 + (lane & 15);
    if (cn >= Nn) continue;
    float bv = bias ? bias[cn] : 0.f;
#pragma unroll
    for (int v = 0; v < 8; v++) {
      int rm = r0 + hh * 8 + v;
      if (rm >= M) continue;
      float val = acc[v] + bv;
      if (act == 1) val = val / (1.f + __expf(-1.702f * val));   // quick_gelu
      size_t orow = rowremap ? ((size_t)(rm / TN_) * L_ + (size_t)(rm % TN_)) : (size_t)rm;
      if (Res)  val += Res[orow * (size_t)Nn + cn];
      if (Of32) Of32[orow * (size_t)Nn + cn] = val;
      if (Of16) Of16[orow * (size_t)Nn + cn] = (f16)val;
    }
  }
}

// ---------------- attention ----------------
// grid (QTILES_, H_, BT_), block 128 (4 waves)
__global__ void attn_kernel(const f16* __restrict__ Q, const f16* __restrict__ Kf,
                            const f16* __restrict__ Vp, const float* __restrict__ bias_table,
                            f16* __restrict__ Oattn, float* __restrict__ clsbuf) {
  __shared__ float S[16 * KP_];
  __shared__ f16   P[16 * KP_];
  int qtile = blockIdx.x, h = blockIdx.y, bt = blockIdx.z;
  int b = bt / T_, t = bt % T_;
  int tid = threadIdx.x, lane = tid & 31, wv = tid >> 5;
  int hh = lane >> 4;
  float b0 = bias_table[h * 2 + 0], b1 = bias_table[h * 2 + 1];
  size_t baseRow = (size_t)b * L_;

  // ---- scores: S = 0.125 * q @ k^T + bias, wmma over HD in 2 chunks ----
  int qi = qtile * 16 + (lane & 15); if (qi > N_ - 1) qi = N_ - 1;
  const f16* qrow = Q + (baseRow + (size_t)t * N_ + qi) * D_ + h * HD_;
  for (int j = wv; j < 19; j += 4) {
    v8f acc = {};
    int key = j * 16 + (lane & 15);
    size_t kr = 0; bool kok = (key < KEYS_);
    if (kok) kr = baseRow + (key < N_ ? (size_t)t * N_ + key : (size_t)TN_ + (key - N_));
#pragma unroll
    for (int ch = 0; ch < 2; ch++) {
      v16h a, bf;
      const f16* pa = qrow + ch * 32;
#pragma unroll
      for (int e = 0; e < 8; e++) a[e]     = pa[hh * 16 + e];
#pragma unroll
      for (int e = 0; e < 8; e++) a[8 + e] = pa[hh * 16 + 8 + e];
      if (kok) {
        const f16* pb = Kf + kr * D_ + h * HD_ + ch * 32 + hh * 16;
#pragma unroll
        for (int e = 0; e < 16; e++) bf[e] = pb[e];
      } else {
#pragma unroll
        for (int e = 0; e < 16; e++) bf[e] = (f16)0.f;
      }
      acc = wmma32(a, bf, acc);
    }
#pragma unroll
    for (int v = 0; v < 8; v++) {
      int rr = hh * 8 + v;
      float val = acc[v] * 0.125f + (key < N_ ? b0 : b1);
      if (key >= KEYS_) val = -1e30f;
      S[rr * KP_ + key] = val;
    }
  }
  __syncthreads();

  // ---- softmax: 16 rows x 8 threads each (wave32 shuffles) ----
  int row = tid >> 3, sl = tid & 7;
  float m = -1e30f;
  for (int key = sl; key < KEYS_; key += 8) m = fmaxf(m, S[row * KP_ + key]);
  m = fmaxf(m, __shfl_xor(m, 1, 32));
  m = fmaxf(m, __shfl_xor(m, 2, 32));
  m = fmaxf(m, __shfl_xor(m, 4, 32));
  float sum = 0.f;
  for (int key = sl; key < KEYS_; key += 8) {
    float e = __expf(S[row * KP_ + key] - m);
    S[row * KP_ + key] = e; sum += e;
  }
  sum += __shfl_xor(sum, 1, 32);
  sum += __shfl_xor(sum, 2, 32);
  sum += __shfl_xor(sum, 4, 32);
  float inv = 1.f / sum;
  for (int key = sl; key < KP_; key += 8) {
    float p = (key < KEYS_) ? S[row * KP_ + key] * inv : 0.f;
    P[row * KP_ + key] = (f16)p;
    if (qtile == 0 && row == 0 && key >= 1 && key < N_)
      clsbuf[((size_t)bt * H_ + h) * NP_ + (key - 1)] = p;
  }
  __syncthreads();

  // ---- O = P @ V ; wave wv handles hd tile n0 = wv*16 (packed V fragments) ----
  int n0 = wv * 16;
  int nn = n0 + (lane & 15);
  v8f oacc = {};
  size_t vbase = (((size_t)bt * H_ + h) * KCH_) * 64 * 32;
  for (int ch = 0; ch < KCH_; ch++) {
    v16h a, bf;
    const f16* pp = P + (lane & 15) * KP_ + ch * 32;
#pragma unroll
    for (int e = 0; e < 8; e++) a[e]     = pp[hh * 16 + e];
#pragma unroll
    for (int e = 0; e < 8; e++) a[8 + e] = pp[hh * 16 + 8 + e];
    const f16* pv = Vp + vbase + ((size_t)ch * 64 + nn) * 32 + hh * 16;
#pragma unroll
    for (int e = 0; e < 16; e++) bf[e] = pv[e];
    oacc = wmma32(a, bf, oacc);
  }
#pragma unroll
  for (int v = 0; v < 8; v++) {
    int q = qtile * 16 + hh * 8 + v;
    if (q < N_) Oattn[((size_t)bt * N_ + q) * D_ + h * HD_ + nn] = (f16)oacc[v];
  }
}

// ---------------- CLS top-k ----------------
__global__ void topk_kernel(const float* __restrict__ clsbuf, int* __restrict__ idxout) {
  int bt = blockIdx.x;
  __shared__ float m[NP_];
  __shared__ int used[NP_];
  int tid = threadIdx.x;
  for (int p = tid; p < NP_; p += blockDim.x) {
    float s = 0.f;
    for (int h = 0; h < H_; h++) s += clsbuf[((size_t)bt * H_ + h) * NP_ + p];
    m[p] = s * (1.f / H_);
    used[p] = 0;
  }
  __syncthreads();
  if (tid == 0) {
    for (int k = 0; k < TOPS_; k++) {
      float best = -1e30f; int bi = 0;
      for (int p = 0; p < NP_; p++)
        if (!used[p] && m[p] > best) { best = m[p]; bi = p; }
      used[bi] = 1;
      idxout[bt * TOPS_ + k] = bi;
    }
  }
}

// ---------------- gather seeds ----------------
__global__ void gather_seed_kernel(const float* __restrict__ xf, const int* __restrict__ idx,
                                   float* __restrict__ seed, float* __restrict__ size) {
  int b = blockIdx.y, s = blockIdx.x;
  int t = s / TOPS_, mth = s % TOPS_;
  int jp = idx[(b * T_ + t) * TOPS_ + mth];
  const float* src = xf + ((size_t)b * L_ + (size_t)t * N_ + 1 + jp) * D_;
  float* dst = seed + ((size_t)b * SEED0_ + s) * D_;
  for (int c = threadIdx.x; c < D_; c += blockDim.x) dst[c] = src[c];
  if (threadIdx.x == 0) size[b * SEED0_ + s] = 1.f;
}

// ---------------- ToMe: row inverse norms ----------------
__global__ void row_norm_kernel(const float* __restrict__ seed, float* __restrict__ norms) {
  int b = blockIdx.y, r = blockIdx.x;
  const float* p = seed + ((size_t)b * SEED0_ + r) * D_;
  __shared__ float sd[256];
  float s = 0.f;
  for (int c = threadIdx.x; c < D_; c += 256) { float v = p[c]; s += v * v; }
  sd[threadIdx.x] = s; __syncthreads();
  for (int o = 128; o > 0; o >>= 1) { if (threadIdx.x < o) sd[threadIdx.x] += sd[threadIdx.x + o]; __syncthreads(); }
  if (threadIdx.x == 0) norms[b * SEED0_ + r] = rsqrtf(sd[0] + 1e-12f);
}

// ---------------- ToMe: scores + row argmax ----------------
__global__ void tome_scores_kernel(const float* __restrict__ seed, const float* __restrict__ norms,
                                   float* __restrict__ node_max, int* __restrict__ node_idx, int na) {
  int b = blockIdx.y, i = blockIdx.x;
  __shared__ float alds[D_];
  __shared__ float bval[256];
  __shared__ int   bidx[256];
  float an = norms[b * SEED0_ + 2 * i];
  const float* ap = seed + ((size_t)b * SEED0_ + 2 * i) * D_;
  for (int c = threadIdx.x; c < D_; c += 256) alds[c] = ap[c] * an;
  __syncthreads();
  float best = -1e30f; int bi = 0x7fffffff;
  for (int j = threadIdx.x; j < na; j += 256) {
    const float* bp = seed + ((size_t)b * SEED0_ + 2 * j + 1) * D_;
    float dot = 0.f;
    for (int c = 0; c < D_; c++) dot += alds[c] * bp[c];
    dot *= norms[b * SEED0_ + 2 * j + 1];
    if (dot > best || (dot == best && j < bi)) { best = dot; bi = j; }
  }
  bval[threadIdx.x] = best; bidx[threadIdx.x] = bi; __syncthreads();
  for (int o = 128; o > 0; o >>= 1) {
    if (threadIdx.x < o) {
      float v2 = bval[threadIdx.x + o]; int i2 = bidx[threadIdx.x + o];
      if (v2 > bval[threadIdx.x] || (v2 == bval[threadIdx.x] && i2 < bidx[threadIdx.x])) {
        bval[threadIdx.x] = v2; bidx[threadIdx.x] = i2;
      }
    }
    __syncthreads();
  }
  if (threadIdx.x == 0) {
    node_max[b * NA_MAX_ + i] = bval[0];
    node_idx[b * NA_MAX_ + i] = bidx[0];
  }
}

// ---------------- ToMe: bitonic argsort (desc value, asc index) ----------------
__global__ void tome_sort_kernel(const float* __restrict__ node_max, int* __restrict__ edge_idx, int na) {
  int b = blockIdx.x, tid = threadIdx.x;
  __shared__ float val[512];
  __shared__ int   idx[512];
  val[tid] = (tid < na) ? node_max[b * NA_MAX_ + tid] : -3e38f;
  idx[tid] = (tid < na) ? tid : (0x40000000 + tid);
  for (int k = 2; k <= 512; k <<= 1)
    for (int j = k >> 1; j > 0; j >>= 1) {
      __syncthreads();
      int ixj = tid ^ j;
      if (ixj > tid) {
        bool up = ((tid & k) == 0);
        float va = val[tid], vb = val[ixj];
        int ia = idx[tid], ib = idx[ixj];
        bool before = (va > vb) || (va == vb && ia < ib);
        if (up ? !before : before) { val[tid] = vb; val[ixj] = va; idx[tid] = ib; idx[ixj] = ia; }
      }
    }
  __syncthreads();
  if (tid < na) edge_idx[b * NA_MAX_ + tid] = idx[tid];
}

// ---------------- ToMe: copy unm + dst rows (times size) ----------------
__global__ void tome_merge_copy_kernel(const float* __restrict__ seedIn, const float* __restrict__ sizeIn,
                                       const int* __restrict__ edge_idx, int na, int r,
                                       float* __restrict__ seedOut, float* __restrict__ sizeOut) {
  int b = blockIdx.y, o = blockIdx.x;
  int unm = na - r;
  int inRow = (o < unm) ? 2 * edge_idx[b * NA_MAX_ + r + o] : 2 * (o - unm) + 1;
  float w = sizeIn[b * SEED0_ + inRow];
  const float* sp = seedIn + ((size_t)b * SEED0_ + inRow) * D_;
  float* dp = seedOut + ((size_t)b * SEED0_ + o) * D_;
  for (int c = threadIdx.x; c < D_; c += blockDim.x) dp[c] = sp[c] * w;
  if (threadIdx.x == 0) sizeOut[b * SEED0_ + o] = w;
}

// ---------------- ToMe: scatter-add src rows into dst ----------------
__global__ void tome_merge_add_kernel(const float* __restrict__ seedIn, const float* __restrict__ sizeIn,
                                      const int* __restrict__ edge_idx, const int* __restrict__ node_idx,
                                      int na, int r, float* __restrict__ seedOut, float* __restrict__ sizeOut) {
  int b = blockIdx.y, s = blockIdx.x;
  int si = edge_idx[b * NA_MAX_ + s];
  int dj = node_idx[b * NA_MAX_ + si];
  int inRow = 2 * si;
  int oRow = (na - r) + dj;
  float w = sizeIn[b * SEED0_ + inRow];
  const float* sp = seedIn + ((size_t)b * SEED0_ + inRow) * D_;
  float* dp = seedOut + ((size_t)b * SEED0_ + oRow) * D_;
  for (int c = threadIdx.x; c < D_; c += blockDim.x) atomicAdd(&dp[c], sp[c] * w);
  if (threadIdx.x == 0) atomicAdd(&sizeOut[b * SEED0_ + oRow], w);
}

// ---------------- ToMe: divide by size ----------------
__global__ void tome_div_kernel(float* __restrict__ seed, const float* __restrict__ size) {
  int b = blockIdx.y, o = blockIdx.x;
  float inv = 1.f / size[b * SEED0_ + o];
  float* p = seed + ((size_t)b * SEED0_ + o) * D_;
  for (int c = threadIdx.x; c < D_; c += blockDim.x) p[c] *= inv;
}

// ---------------- append final seeds to xf ----------------
__global__ void seed_to_xf_kernel(const float* __restrict__ seed, float* __restrict__ xf) {
  int b = blockIdx.y, s = blockIdx.x;
  const float* sp = seed + ((size_t)b * SEED0_ + s) * D_;
  float* dp = xf + ((size_t)b * L_ + TN_ + s) * D_;
  for (int c = threadIdx.x; c < D_; c += blockDim.x) dp[c] = sp[c];
}

// =========================== host launcher ===========================
extern "C" void kernel_launch(void* const* d_in, const int* in_sizes, int n_in,
                              void* d_out, int out_size, void* d_ws, size_t ws_size,
                              hipStream_t stream) {
  (void)in_sizes; (void)n_in; (void)out_size; (void)ws_size;
  const float* x     = (const float*)d_in[0];
  const float* wq    = (const float*)d_in[1];
  const float* bq    = (const float*)d_in[2];
  const float* wk    = (const float*)d_in[3];
  const float* bk    = (const float*)d_in[4];
  const float* wv    = (const float*)d_in[5];
  const float* bvb   = (const float*)d_in[6];
  const float* wo    = (const float*)d_in[7];
  const float* bo    = (const float*)d_in[8];
  const float* btab  = (const float*)d_in[9];
  const float* ln1g  = (const float*)d_in[10];
  const float* ln1b  = (const float*)d_in[11];
  const float* ln2g  = (const float*)d_in[12];
  const float* ln2b  = (const float*)d_in[13];
  const float* wfc   = (const float*)d_in[14];
  const float* bfc   = (const float*)d_in[15];
  const float* wproj = (const float*)d_in[16];
  const float* bproj = (const float*)d_in[17];
  float* out = (float*)d_out;

  char* ws = (char*)d_ws;
  size_t off = 0;
  auto alloc = [&](size_t bytes) -> void* {
    void* p = ws + off;
    off += (bytes + 255) & ~(size_t)255;
    return p;
  };

  f16* xn16    = (f16*)alloc((size_t)BL_ * D_ * sizeof(f16));
  f16* q16     = (f16*)alloc((size_t)BL_ * D_ * sizeof(f16));
  f16* k16     = (f16*)alloc((size_t)BL_ * D_ * sizeof(f16));
  f16* v16     = (f16*)alloc((size_t)BL_ * D_ * sizeof(f16));
  f16* attn16  = (f16*)alloc((size_t)BT_ * N_ * D_ * sizeof(f16));
  f16* xfn16   = (f16*)alloc((size_t)BL_ * D_ * sizeof(f16));
  f16* h16     = (f16*)alloc((size_t)BL_ * FF_ * sizeof(f16));
  float* xf    = (float*)alloc((size_t)BL_ * D_ * sizeof(float));
  f16* wq16p    = (f16*)alloc((size_t)D_ * D_ * sizeof(f16));
  f16* wk16p    = (f16*)alloc((size_t)D_ * D_ * sizeof(f16));
  f16* wv16p    = (f16*)alloc((size_t)D_ * D_ * sizeof(f16));
  f16* wo16p    = (f16*)alloc((size_t)D_ * D_ * sizeof(f16));
  f16* wfc16p   = (f16*)alloc((size_t)D_ * FF_ * sizeof(f16));
  f16* wproj16p = (f16*)alloc((size_t)FF_ * D_ * sizeof(f16));
  f16* vpack    = (f16*)alloc((size_t)BT_ * H_ * KCH_ * 64 * 32 * sizeof(f16));
  float* cls   = (float*)alloc((size_t)BT_ * H_ * NP_ * sizeof(float));
  int*   idxs  = (int*)alloc((size_t)BT_ * TOPS_ * sizeof(int));
  float* seedA = (float*)alloc((size_t)B_ * SEED0_ * D_ * sizeof(float));
  float* seedB = (float*)alloc((size_t)B_ * SEED0_ * D_ * sizeof(float));
  float* sizeA = (float*)alloc((size_t)B_ * SEED0_ * sizeof(float));
  float* sizeB = (float*)alloc((size_t)B_ * SEED0_ * sizeof(float));
  float* norms = (float*)alloc((size_t)B_ * SEED0_ * sizeof(float));
  float* nmax  = (float*)alloc((size_t)B_ * NA_MAX_ * sizeof(float));
  int*   nidx  = (int*)alloc((size_t)B_ * NA_MAX_ * sizeof(int));
  int*   eidx  = (int*)alloc((size_t)B_ * NA_MAX_ * sizeof(int));

  // 1) LN1 -> f16
  ln_f16_kernel<<<BL_, 256, 0, stream>>>(x, ln1g, ln1b, xn16, BL_);

  // 2) weights -> packed f16 B-fragment layout
  auto packw = [&](const float* s, f16* d, int K, int Nn) {
    size_t n = (size_t)K * Nn;
    pack_w_kernel<<<(unsigned)((n + 255) / 256), 256, 0, stream>>>(s, d, K, Nn);
  };
  packw(wq, wq16p, D_, D_);    packw(wk, wk16p, D_, D_);
  packw(wv, wv16p, D_, D_);    packw(wo, wo16p, D_, D_);
  packw(wfc, wfc16p, D_, FF_); packw(wproj, wproj16p, FF_, D_);

  // 3) q/k/v projections (WMMA, f16 out)
  dim3 gP(D_ / 256, BL_ / 16);
  gemm_wmma_kernel<<<gP, 128, 0, stream>>>(xn16, wq16p, bq,  nullptr, nullptr, q16, BL_, D_, D_, 0, 0);
  gemm_wmma_kernel<<<gP, 128, 0, stream>>>(xn16, wk16p, bk,  nullptr, nullptr, k16, BL_, D_, D_, 0, 0);
  gemm_wmma_kernel<<<gP, 128, 0, stream>>>(xn16, wv16p, bvb, nullptr, nullptr, v16, BL_, D_, D_, 0, 0);

  // 3b) pack V into per-(bt,h,chunk) fragment layout
  {
    size_t n = (size_t)BT_ * H_ * KCH_ * 64 * 32;
    pack_v_kernel<<<(unsigned)((n + 255) / 256), 256, 0, stream>>>(v16, vpack);
  }

  // 4) attention (WMMA QK^T + WMMA PV)
  attn_kernel<<<dim3(QTILES_, H_, BT_), 128, 0, stream>>>(q16, k16, vpack, btab, attn16, cls);

  // 5) output projection + residual, write into xf[:, :TN] (remapped rows)
  gemm_wmma_kernel<<<dim3(D_ / 256, (BT_ * N_) / 16), 128, 0, stream>>>(
      attn16, wo16p, bo, x, xf, nullptr, BT_ * N_, D_, D_, 0, 1);

  // 6) CLS top-k + seed gather
  topk_kernel<<<BT_, 256, 0, stream>>>(cls, idxs);
  gather_seed_kernel<<<dim3(SEED0_, B_), 256, 0, stream>>>(xf, idxs, seedA, sizeA);

  // 7) ToMe merge schedule
  const int sched[9] = {100, 100, 100, 100, 100, 100, 100, 100, 32};
  float* sCur = seedA; float* zCur = sizeA;
  float* sNxt = seedB; float* zNxt = sizeB;
  int n = SEED0_;
  for (int it = 0; it < 9; it++) {
    int r = sched[it];
    int na = n / 2;
    row_norm_kernel<<<dim3(n, B_), 256, 0, stream>>>(sCur, norms);
    tome_scores_kernel<<<dim3(na, B_), 256, 0, stream>>>(sCur, norms, nmax, nidx, na);
    tome_sort_kernel<<<B_, 512, 0, stream>>>(nmax, eidx, na);
    tome_merge_copy_kernel<<<dim3(n - r, B_), 256, 0, stream>>>(sCur, zCur, eidx, na, r, sNxt, zNxt);
    tome_merge_add_kernel<<<dim3(r, B_), 256, 0, stream>>>(sCur, zCur, eidx, nidx, na, r, sNxt, zNxt);
    tome_div_kernel<<<dim3(n - r, B_), 256, 0, stream>>>(sNxt, zNxt);
    n -= r;
    float* tmp = sCur; sCur = sNxt; sNxt = tmp;
    tmp = zCur; zCur = zNxt; zNxt = tmp;
  }

  // 8) append seeds (n == 96 == KCTX_)
  seed_to_xf_kernel<<<dim3(KCTX_, B_), 256, 0, stream>>>(sCur, xf);

  // 9) MLP: LN2 -> fc (gelu fused, f16 out) -> proj (+bias +xf residual) -> d_out
  ln_f16_kernel<<<BL_, 256, 0, stream>>>(xf, ln2g, ln2b, xfn16, BL_);
  gemm_wmma_kernel<<<dim3(FF_ / 256, BL_ / 16), 128, 0, stream>>>(
      xfn16, wfc16p, bfc, nullptr, nullptr, h16, BL_, FF_, D_, 1, 0);
  gemm_wmma_kernel<<<dim3(D_ / 256, BL_ / 16), 128, 0, stream>>>(
      h16, wproj16p, bproj, xf, out, nullptr, BL_, D_, FF_, 0, 0);
}
